// SwinTransformerBlock1D_88218628260827
// MI455X (gfx1250) — matrix operand order repaired and checked by reading, not compiled, not run on hardware
//
#include <hip/hip_runtime.h>
#include <hip/hip_bf16.h>
#include <math.h>

// ---------------------------------------------------------------------------
// Swin-1D block for MI455X (gfx1250): bf16 WMMA GEMMs + fused windowed attn.
// GEMM tiles 64x128 (8 waves, each wave 16x64 of output -> 4 WMMA tiles),
// BLOCK_K=64 (two v_wmma_f32_16x16x32_bf16 K-steps per LDS stage).
// ---------------------------------------------------------------------------

typedef __bf16 bf16_t;
typedef __attribute__((ext_vector_type(16))) __bf16 v16bf;
typedef __attribute__((ext_vector_type(8)))  float  v8f;

#define DIM      512
#define SEQ      8192
#define BATCH    8
#define NTOK     (BATCH*SEQ)   /* 65536 */
#define WIN      64
#define SHIFT    32
#define HEADS    8
#define HEAD_DIM 64
#define LN_EPS   1e-5f

// -------- WMMA fragment loaders (LDS row stride fixed at 72 bf16 elems) ----
// A (16x32, MxK) per ISA: lane<16 holds row M=lane, K in {kb..kb+7, kb+16..kb+23}
// with kb = (lane>>4)*8. Two ds_load_b128 per fragment.
__device__ __forceinline__ v16bf frag_a(const bf16_t* base, int lane) {
  const int row = lane & 15;
  const int kb  = (lane >> 4) << 3;          // 0 or 8
  union { v16bf v; uint4 q[2]; } u;
  u.q[0] = *(const uint4*)(base + row*72 + kb);
  u.q[1] = *(const uint4*)(base + row*72 + kb + 16);
  return u.v;
}
// B (32x16, KxN), staged transposed in LDS as Bt[n][k]: lane holds col N=lane&15,
// K in {kb..kb+7, kb+16..kb+23} with kb = (lane>>4)*8 (matches A) -> two b128.
__device__ __forceinline__ v16bf frag_b(const bf16_t* base, int lane) {
  const int col = lane & 15;
  const int kb  = (lane >> 4) << 3;          // 0 or 8
  union { v16bf v; uint4 q[2]; } u;
  u.q[0] = *(const uint4*)(base + col*72 + kb);
  u.q[1] = *(const uint4*)(base + col*72 + kb + 16);
  return u.v;
}
__device__ __forceinline__ v8f wmma_bf16(v16bf a, v16bf b, v8f c) {
  return __builtin_amdgcn_wmma_f32_16x16x32_bf16(false, a, false, b, (short)0, c,
                                                 false, false);
}

// D-matrix element (r, lane) -> M = (lane>>4)*8 + r, N = lane&15 (ISA C/D layout)

// ---------------------------------------------------------------------------
// 64x128-tile bf16 GEMM, 256 threads (8 waves). Wave w: rows tm=w&3 (16-strip),
// cols (w>>2)*64 .. +63 (four 16-strips). Per K-step: 1 A frag + 4 B frags,
// 4 WMMAs (2.5 b128 loads per WMMA, A reused 4x).
// EPI: 0 bias->bf16 | 1 bias+GELU(exact)->bf16 | 2 bias+residual->f32
//      3 proj: bias + shift-mask + roll(+SHIFT) scatter -> f32
// ---------------------------------------------------------------------------
template<int EPI>
__global__ __launch_bounds__(256)
void gemm64(const bf16_t* __restrict__ A, const bf16_t* __restrict__ Bw,
            const float* __restrict__ bias, const float* __restrict__ resid,
            void* __restrict__ outp, int M, int N, int K)
{
  __shared__ bf16_t As[64*72];     //  9.2 KB
  __shared__ bf16_t Bt[128*72];    // 18.4 KB
  const int tid  = threadIdx.x;
  const int lane = tid & 31;
  const int wid  = tid >> 5;
  const int tm   = wid & 3;            // 16-row strip of the 64-row tile
  const int tn0  = (wid >> 2) << 2;    // first of four 16-col strips
  const int m0   = blockIdx.x * 64;
  const int n0   = blockIdx.y * 128;

  v8f acc[4];
  #pragma unroll
  for (int t = 0; t < 4; ++t) acc[t] = (v8f){};

  for (int k0 = 0; k0 < K; k0 += 64) {
    // stage A tile 64x64 (row major), 8 bf16 per chunk
    for (int ch = tid; ch < 512; ch += 256) {
      int r = ch >> 3, c = (ch & 7) << 3;
      *(uint4*)&As[r*72 + c] = *(const uint4*)&A[(size_t)(m0 + r)*K + k0 + c];
    }
    // stage B tile 64x128 transposed -> Bt[n][k]
    for (int ch = tid; ch < 1024; ch += 256) {
      int kr = ch >> 4, c = (ch & 15) << 3;
      uint4 raw = *(const uint4*)&Bw[(size_t)(k0 + kr)*N + n0 + c];
      const bf16_t* e = (const bf16_t*)&raw;
      #pragma unroll
      for (int j = 0; j < 8; ++j) Bt[(c + j)*72 + kr] = e[j];
    }
    if (k0 + 64 < K) {   // pull next tiles toward L2 (global_prefetch_b8)
      __builtin_prefetch(&A[(size_t)(m0 + (tid >> 2))*K + k0 + 64], 0, 1);
      __builtin_prefetch(&Bw[(size_t)(k0 + 64 + (tid >> 2))*N + n0 + ((tid & 3) << 5)], 0, 1);
    }
    __syncthreads();
    #pragma unroll
    for (int kk = 0; kk < 64; kk += 32) {
      v16bf a = frag_a(As + (tm*16)*72 + kk, lane);
      #pragma unroll
      for (int t = 0; t < 4; ++t) {
        v16bf b = frag_b(Bt + ((tn0 + t)*16)*72 + kk, lane);
        acc[t] = wmma_bf16(a, b, acc[t]);
      }
    }
    __syncthreads();
  }

  const int mBase = m0 + tm*16 + ((lane >> 4) << 3);
  #pragma unroll
  for (int t = 0; t < 4; ++t) {
    const int nB = n0 + (tn0 + t)*16 + (lane & 15);
    const float bz = bias[nB];
    #pragma unroll
    for (int r = 0; r < 8; ++r) {
      const int m = mBase + r;
      float v = acc[t][r] + bz;
      if constexpr (EPI == 0) {
        ((bf16_t*)outp)[(size_t)m*N + nB] = (bf16_t)v;
      } else if constexpr (EPI == 1) {
        ((bf16_t*)outp)[(size_t)m*N + nB] =
            (bf16_t)(0.5f*v*(1.0f + erff(v*0.7071067811865475f)));
      } else if constexpr (EPI == 2) {
        ((float*)outp)[(size_t)m*N + nB] = v + resid[(size_t)m*N + nB];
      } else { // proj: row m in shifted domain; mask boundary rows, roll back
        const int b = m / SEQ, p = m % SEQ;
        const bool msk = (p < SHIFT) | (p >= SEQ - SHIFT);
        const int l = (p + SHIFT) & (SEQ - 1);
        ((float*)outp)[((size_t)b*SEQ + l)*(size_t)N + nB] = msk ? 0.0f : v;
      }
    }
  }
}

// ---------------------------------------------------------------------------
// Windowed attention: one block per (window, head). WMMA for QK^T and PV.
// ---------------------------------------------------------------------------
__global__ __launch_bounds__(256)
void attn_win(const bf16_t* __restrict__ qkv, const float* __restrict__ rbt,
              bf16_t* __restrict__ out)
{
  __shared__ bf16_t Qs[64*72];
  __shared__ bf16_t Ks[64*72];   // K row-major == Bt layout for S = Q K^T
  __shared__ bf16_t Vt[64*72];   // V transposed: Vt[d][key]
  __shared__ bf16_t Ps[64*72];
  __shared__ float  Ss[64*66];

  const int tid  = threadIdx.x;
  const int lane = tid & 31;
  const int wid  = tid >> 5;
  const int win  = blockIdx.x;
  const int head = blockIdx.y;
  const size_t rowStride = 3*DIM;
  const size_t base = (size_t)win * WIN;

  { // stage Q, K, V^T: thread -> row i = tid/4, 16 cols at c0
    const int i = tid >> 2, c0 = (tid & 3) << 4;
    const bf16_t* qrow = qkv + (base + i)*rowStride + head*HEAD_DIM;
    *(uint4*)&Qs[i*72 + c0]     = *(const uint4*)(qrow + c0);
    *(uint4*)&Qs[i*72 + c0 + 8] = *(const uint4*)(qrow + c0 + 8);
    const bf16_t* krow = qrow + DIM;
    *(uint4*)&Ks[i*72 + c0]     = *(const uint4*)(krow + c0);
    *(uint4*)&Ks[i*72 + c0 + 8] = *(const uint4*)(krow + c0 + 8);
    const bf16_t* vrow = qrow + 2*DIM;
    uint4 v0 = *(const uint4*)(vrow + c0);
    uint4 v1 = *(const uint4*)(vrow + c0 + 8);
    const bf16_t* e0 = (const bf16_t*)&v0;
    const bf16_t* e1 = (const bf16_t*)&v1;
    #pragma unroll
    for (int j = 0; j < 8; ++j) { Vt[(c0+j)*72 + i] = e0[j]; Vt[(c0+8+j)*72 + i] = e1[j]; }
  }
  __syncthreads();

  // S = Q K^T : 16 tiles of 16x16, two per wave
  #pragma unroll
  for (int t = 0; t < 2; ++t) {
    const int tt = wid*2 + t, tmq = tt >> 2, tnk = tt & 3;
    v8f acc = {};
    #pragma unroll
    for (int kk = 0; kk < 64; kk += 32) {
      v16bf a = frag_a(Qs + tmq*16*72 + kk, lane);
      v16bf b = frag_b(Ks + tnk*16*72 + kk, lane);
      acc = wmma_bf16(a, b, acc);
    }
    const int rB = tmq*16 + ((lane>>4)<<3), cB = tnk*16 + (lane & 15);
    #pragma unroll
    for (int r = 0; r < 8; ++r) Ss[(rB + r)*66 + cB] = acc[r];
  }
  __syncthreads();

  // softmax(S*scale + rel_bias) per row; P in bf16
  if (tid < 64) {
    const int row = tid;
    float mx = -3.4e38f;
    for (int j = 0; j < 64; ++j) {
      float v = Ss[row*66 + j]*0.125f + rbt[(row - j + WIN - 1)*HEADS + head];
      Ss[row*66 + j] = v;
      mx = fmaxf(mx, v);
    }
    float sum = 0.0f;
    for (int j = 0; j < 64; ++j) {
      float e = __expf(Ss[row*66 + j] - mx);
      Ss[row*66 + j] = e;
      sum += e;
    }
    const float inv = 1.0f / sum;
    for (int j = 0; j < 64; ++j) Ps[row*72 + j] = (bf16_t)(Ss[row*66 + j]*inv);
  }
  __syncthreads();

  // O = P V  -> out channel = head*64 + d
  #pragma unroll
  for (int t = 0; t < 2; ++t) {
    const int tt = wid*2 + t, tmq = tt >> 2, tnd = tt & 3;
    v8f acc = {};
    #pragma unroll
    for (int kk = 0; kk < 64; kk += 32) {
      v16bf a = frag_a(Ps + tmq*16*72 + kk, lane);
      v16bf b = frag_b(Vt + tnd*16*72 + kk, lane);
      acc = wmma_bf16(a, b, acc);
    }
    const int rB = tmq*16 + ((lane>>4)<<3), cB = tnd*16 + (lane & 15);
    #pragma unroll
    for (int r = 0; r < 8; ++r)
      out[(base + rB + r)*DIM + head*HEAD_DIM + cB] = (bf16_t)acc[r];
  }
}

// ---------------------------------------------------------------------------
// LN1 + cyclic shift(-SHIFT), fp32 -> bf16 (one block per token)
// ---------------------------------------------------------------------------
__global__ __launch_bounds__(256)
void ln1_shift(const float* __restrict__ x, const float* __restrict__ g,
               const float* __restrict__ bb, bf16_t* __restrict__ out)
{
  __shared__ float s1[256], s2[256];
  const int row = blockIdx.x;
  const int tid = threadIdx.x;
  const size_t rbase = (size_t)row * DIM;
  const float xa = x[rbase + tid];
  const float xb = x[rbase + tid + 256];
  s1[tid] = xa + xb; s2[tid] = xa*xa + xb*xb;
  __syncthreads();
  for (int s = 128; s > 0; s >>= 1) {
    if (tid < s) { s1[tid] += s1[tid+s]; s2[tid] += s2[tid+s]; }
    __syncthreads();
  }
  const float mu  = s1[0] * (1.0f/DIM);
  const float var = s2[0] * (1.0f/DIM) - mu*mu;
  const float rs  = rsqrtf(var + LN_EPS);
  const int b = row / SEQ, l = row % SEQ;
  const size_t drow = ((size_t)b*SEQ + ((l + SEQ - SHIFT) & (SEQ-1))) * DIM;
  out[drow + tid]       = (bf16_t)((xa - mu)*rs*g[tid]     + bb[tid]);
  out[drow + tid + 256] = (bf16_t)((xb - mu)*rs*g[tid+256] + bb[tid+256]);
}

// ---------------------------------------------------------------------------
// x2 = x + h_proj (fp32 kept for final residual), ln2 = LN(x2) -> bf16
// ---------------------------------------------------------------------------
__global__ __launch_bounds__(256)
void resid_ln2(const float* __restrict__ x, const float* __restrict__ h,
               const float* __restrict__ g, const float* __restrict__ bb,
               float* __restrict__ x2, bf16_t* __restrict__ ln2)
{
  __shared__ float s1[256], s2[256];
  const int row = blockIdx.x;
  const int tid = threadIdx.x;
  const size_t rbase = (size_t)row * DIM;
  const float xa = x[rbase + tid]       + h[rbase + tid];
  const float xb = x[rbase + tid + 256] + h[rbase + tid + 256];
  x2[rbase + tid] = xa; x2[rbase + tid + 256] = xb;
  s1[tid] = xa + xb; s2[tid] = xa*xa + xb*xb;
  __syncthreads();
  for (int s = 128; s > 0; s >>= 1) {
    if (tid < s) { s1[tid] += s1[tid+s]; s2[tid] += s2[tid+s]; }
    __syncthreads();
  }
  const float mu  = s1[0] * (1.0f/DIM);
  const float var = s2[0] * (1.0f/DIM) - mu*mu;
  const float rs  = rsqrtf(var + LN_EPS);
  ln2[rbase + tid]       = (bf16_t)((xa - mu)*rs*g[tid]     + bb[tid]);
  ln2[rbase + tid + 256] = (bf16_t)((xb - mu)*rs*g[tid+256] + bb[tid+256]);
}

__global__ void f32_to_bf16(const float* __restrict__ in, bf16_t* __restrict__ out, int n) {
  int i = blockIdx.x*blockDim.x + threadIdx.x;
  if (i < n) out[i] = (bf16_t)in[i];
}

// ---------------------------------------------------------------------------
extern "C" void kernel_launch(void* const* d_in, const int* in_sizes, int n_in,
                              void* d_out, int out_size, void* d_ws, size_t ws_size,
                              hipStream_t stream)
{
  (void)in_sizes; (void)n_in; (void)out_size; (void)ws_size;
  const float* x      = (const float*)d_in[0];
  const float* ln1_g  = (const float*)d_in[1];
  const float* ln1_b  = (const float*)d_in[2];
  const float* qkv_w  = (const float*)d_in[3];
  const float* qkv_b  = (const float*)d_in[4];
  const float* proj_w = (const float*)d_in[5];
  const float* proj_b = (const float*)d_in[6];
  const float* rbt    = (const float*)d_in[7];
  const float* ln2_g  = (const float*)d_in[8];
  const float* ln2_b  = (const float*)d_in[9];
  const float* m1_w   = (const float*)d_in[10];
  const float* m1_b   = (const float*)d_in[11];
  const float* m2_w   = (const float*)d_in[12];
  const float* m2_b   = (const float*)d_in[13];

  char* ws = (char*)d_ws;
  size_t off = 0;
  auto take = [&](size_t bytes) {
    char* p = ws + off;
    off = (off + bytes + 255) & ~(size_t)255;
    return p;
  };

  bf16_t* w_qkv = (bf16_t*)take((size_t)DIM*3*DIM*2);
  bf16_t* w_prj = (bf16_t*)take((size_t)DIM*DIM*2);
  bf16_t* w_m1  = (bf16_t*)take((size_t)DIM*4*DIM*2);
  bf16_t* w_m2  = (bf16_t*)take((size_t)4*DIM*DIM*2);
  bf16_t* h_sh  = (bf16_t*)take((size_t)NTOK*DIM*2);                       // reused as ln2
  char*   big   = take((size_t)NTOK*3*DIM*2 + (size_t)NTOK*DIM*2);         // qkv+attn / mlp1
  bf16_t* qkvb  = (bf16_t*)big;
  bf16_t* attnb = (bf16_t*)(big + (size_t)NTOK*3*DIM*2);
  bf16_t* mlp1b = (bf16_t*)big;             // reuse: qkv & attn dead by then
  float*  hproj = (float*)take((size_t)NTOK*DIM*4);
  float*  x2    = (float*)take((size_t)NTOK*DIM*4);
  bf16_t* ln2b  = h_sh;                     // reuse: h_sh dead after QKV GEMM

  // weights -> bf16
  f32_to_bf16<<<(DIM*3*DIM+255)/256, 256, 0, stream>>>(qkv_w, w_qkv, DIM*3*DIM);
  f32_to_bf16<<<(DIM*DIM+255)/256,   256, 0, stream>>>(proj_w, w_prj, DIM*DIM);
  f32_to_bf16<<<(DIM*4*DIM+255)/256, 256, 0, stream>>>(m1_w,  w_m1, DIM*4*DIM);
  f32_to_bf16<<<(4*DIM*DIM+255)/256, 256, 0, stream>>>(m2_w,  w_m2, 4*DIM*DIM);

  // LN1 + shift
  ln1_shift<<<NTOK, 256, 0, stream>>>(x, ln1_g, ln1_b, h_sh);

  // QKV GEMM: [65536,512] x [512,1536]
  gemm64<0><<<dim3(NTOK/64, 3*DIM/128), 256, 0, stream>>>(
      h_sh, w_qkv, qkv_b, nullptr, qkvb, NTOK, 3*DIM, DIM);

  // windowed attention (1024 windows x 8 heads)
  attn_win<<<dim3(NTOK/WIN, HEADS), 256, 0, stream>>>(qkvb, rbt, attnb);

  // proj GEMM + mask + roll-back scatter -> fp32 h
  gemm64<3><<<dim3(NTOK/64, DIM/128), 256, 0, stream>>>(
      attnb, w_prj, proj_b, nullptr, hproj, NTOK, DIM, DIM);

  // x2 = x + h ; ln2 = LN(x2)
  resid_ln2<<<NTOK, 256, 0, stream>>>(x, hproj, ln2_g, ln2_b, x2, ln2b);

  // MLP1 GEMM + exact GELU: [65536,512] x [512,2048]
  gemm64<1><<<dim3(NTOK/64, 4*DIM/128), 256, 0, stream>>>(
      ln2b, w_m1, m1_b, nullptr, mlp1b, NTOK, 4*DIM, DIM);

  // MLP2 GEMM + residual -> fp32 output
  gemm64<2><<<dim3(NTOK/64, DIM/128), 256, 0, stream>>>(
      mlp1b, w_m2, m2_b, x2, d_out, NTOK, DIM, 4*DIM);
}
